// TFLongformerSelfAttention_38079180046613
// MI455X (gfx1250) — matrix operand order repaired and checked by reading, not compiled, not run on hardware
//
#include <hip/hip_runtime.h>
#include <hip/hip_bf16.h>

typedef __bf16 bf16_t;
typedef __attribute__((ext_vector_type(16))) __bf16 v16bf;
typedef __attribute__((ext_vector_type(8)))  float  v8f;
typedef unsigned int u32x4 __attribute__((ext_vector_type(4)));
typedef int          i32x8 __attribute__((ext_vector_type(8)));

#define NEGINF (-__builtin_inff())

static constexpr int Bc = 2, Sc = 4096, Ec = 1024, Hc = 16, Dc = 64, Gc = 64;
static constexpr int Wc = 256;                 // one-sided window
static constexpr int LW = 544;                 // padded local strip (17 * 32)
static constexpr int PCOLS = Gc + 2 * Wc + 1;  // 577 prob columns

// ---------------- WMMA helpers (CDNA5 wave32 layouts) ----------------

__device__ __forceinline__ v8f wmma_bf16(v16bf a, v16bf b, v8f c) {
  return __builtin_amdgcn_wmma_f32_16x16x32_bf16(false, a, false, b, (short)0, c,
                                                 false, false);
}

// A fragment: 16x32 bf16, row-major source, row stride ld.
__device__ __forceinline__ v16bf load_a_frag(const bf16_t* p, int ld, int lane) {
  int m  = lane & 15;
  int kb = (lane >> 4) * 16;
  const bf16_t* r = p + (size_t)m * ld + kb;
  v16bf a;
#pragma unroll
  for (int j = 0; j < 16; ++j) a[j] = r[j];
  return a;
}

// B fragment: 32x16 where memory holds B^T row-major: elem(kd,n) at p[n*ld+kd].
__device__ __forceinline__ v16bf load_bt_frag(const bf16_t* p, int ld, int lane) {
  int n  = lane & 15;
  int kb = (lane >> 4) * 16;
  const bf16_t* r = p + (size_t)n * ld + kb;
  v16bf b;
#pragma unroll
  for (int j = 0; j < 16; ++j) b[j] = r[j];
  return b;
}

// ---------------- CDNA5 async / TDM primitives ----------------

// One 16-byte async global->LDS transfer per lane (ASYNCcnt).
__device__ __forceinline__ void async_g2l_b128(unsigned lds_byte_addr,
                                               const void* gaddr) {
  unsigned long long ga = (unsigned long long)gaddr;
  asm volatile("global_load_async_to_lds_b128 %0, %1, off"
               :: "v"(lds_byte_addr), "v"(ga) : "memory");
}
__device__ __forceinline__ void wait_asynccnt0() {
  asm volatile("s_wait_asynccnt 0x0" ::: "memory");
}

// TDM: 2D tile load (bf16 elements), descriptor per CDNA5 ISA.
// tensor_d0/tile_d0: row length (elements); tensor_d1: rows available (OOB rows
// read as zero); tile_d1: rows to move; stride0: tensor row stride (elements).
__device__ __forceinline__ void tdm_load_2d_bf16(unsigned lds_byte_addr,
                                                 const void* gaddr,
                                                 unsigned tensor_d0,
                                                 unsigned tensor_d1,
                                                 unsigned tile_d0,
                                                 unsigned tile_d1,
                                                 unsigned stride0) {
  unsigned long long ga = (unsigned long long)gaddr;
  u32x4 g0;
  g0[0] = 1u;                                         // count=1 (valid), user mode
  g0[1] = lds_byte_addr;                              // lds_addr
  g0[2] = (unsigned)(ga & 0xffffffffu);               // global_addr[31:0]
  g0[3] = (unsigned)((ga >> 32) & 0x1ffffffu) | (2u << 30);  // addr[56:32] | type=2
  i32x8 g1;
  g1[0] = (int)(1u << 16);                            // data_size=1 (2 bytes)
  g1[1] = (int)((tensor_d0 & 0xffffu) << 16);         // td0[15:0] @ bits 63:48
  g1[2] = (int)(((tensor_d0 >> 16) & 0xffffu) | ((tensor_d1 & 0xffffu) << 16));
  g1[3] = (int)(((tensor_d1 >> 16) & 0xffffu) | ((tile_d0 & 0xffffu) << 16));
  g1[4] = (int)(tile_d1 & 0xffffu);                   // tile_dim1 (tile_dim2=0)
  g1[5] = (int)stride0;                               // tensor_dim0_stride[31:0]
  g1[6] = 0;                                          // stride0[47:32] | stride1 lo
  g1[7] = 0;
  asm volatile("tensor_load_to_lds %0, %1" :: "s"(g0), "s"(g1) : "memory");
}

// ---------------- conversion kernels ----------------

__global__ void k_cvt_bf16(const float* __restrict__ in, bf16_t* __restrict__ out, int n) {
  int i = blockIdx.x * 256 + threadIdx.x;
  if (i < n) out[i] = (bf16_t)in[i];
}

__global__ void k_cvt_transpose(const float* __restrict__ in, bf16_t* __restrict__ out) {
  int idx = blockIdx.x * 256 + threadIdx.x;     // over E*E
  int k = idx >> 10, n = idx & (Ec - 1);
  out[(size_t)n * Ec + k] = (bf16_t)in[idx];
}

// ---------------- fused 6-way projection GEMM -------
// block = 128 (4 waves); block tile 16(M) x 256(N); wave tile 16x64 (4 accums).
// A tile (16x1024 bf16, contiguous 32KB) staged via async global->LDS.

struct GemmArgs {
  const float* bias[6];
  float scale[6];
};

__global__ void k_proj_gemm(const bf16_t* __restrict__ hsb,
                            const bf16_t* __restrict__ wtb,
                            bf16_t* __restrict__ proj, GemmArgs ga) {
  int mt   = blockIdx.x;              // 0..511
  int ng   = blockIdx.y;              // 0..3 (256-col groups)
  int w    = blockIdx.z;              // weight index
  int wave = threadIdx.x >> 5;
  int lane = threadIdx.x & 31;
  int nbase = ng * 256 + wave * 64;

  __shared__ bf16_t atile[16 * Ec];   // 32 KB

  const bf16_t* wt   = wtb  + (size_t)w * Ec * Ec;
  bf16_t*       out  = proj + (size_t)w * (Bc * Sc) * Ec;
  const bf16_t* asrc = hsb  + (size_t)mt * 16 * Ec;   // 16 contiguous rows

  // stage A: 2048 x b128 async transfers
  unsigned abase = (unsigned)(uintptr_t)(void*)atile;
  for (int i = threadIdx.x; i < (16 * Ec) / 8; i += 128)
    async_g2l_b128(abase + (unsigned)(i * 16), asrc + (size_t)i * 8);
  wait_asynccnt0();
  __syncthreads();

  v8f acc[4] = {v8f{}, v8f{}, v8f{}, v8f{}};
  for (int ks = 0; ks < Ec; ks += 32) {
    v16bf a = load_a_frag(atile + ks, Ec, lane);      // LDS
#pragma unroll
    for (int s = 0; s < 4; ++s) {
      const bf16_t* bp = wt + (size_t)(nbase + s * 16) * Ec + ks;
      __builtin_prefetch(bp + 128, 0, 1);
      v16bf b = load_bt_frag(bp, Ec, lane);
      acc[s] = wmma_bf16(a, b, acc[s]);
    }
  }
  int mo = (lane >> 4) * 8;
  float sc = ga.scale[w];
#pragma unroll
  for (int s = 0; s < 4; ++s) {
    int n = nbase + s * 16 + (lane & 15);
    float bv = ga.bias[w][n];
#pragma unroll
    for (int r = 0; r < 8; ++r)
      out[(size_t)(mt * 16 + mo + r) * Ec + n] = (bf16_t)((acc[s][r] + bv) * sc);
  }
}

// ---------------- sliding-window attention ----------
// block = 128 (4 waves), one (b, h, 16-query tile); K/V strips staged via TDM.
__global__ void k_sliding_attn(const bf16_t* __restrict__ qb,
                               const bf16_t* __restrict__ kb,
                               const bf16_t* __restrict__ vb,
                               float* __restrict__ out_att,
                               float* __restrict__ out_probs) {
  int qt = blockIdx.x, h = blockIdx.y, bb = blockIdx.z;
  int qbase = qt * 16;
  int jbase = qbase - Wc;
  int tid = threadIdx.x, wave = tid >> 5, lane = tid & 31;

  extern __shared__ char smem[];
  float*  s_loc  = (float*)smem;                                  // 16 x LW f32
  float*  s_glob = (float*)(smem + 16 * LW * 4);                  // 16 x 64 f32
  bf16_t* p_loc  = (bf16_t*)(smem + 16 * LW * 4 + 16 * 64 * 4);   // 16 x LW bf16
  bf16_t* p_glob = p_loc + 16 * LW;                               // 16 x 64 bf16
  bf16_t* kst    = (bf16_t*)(smem + 58368);                       // LW x 64 bf16
  bf16_t* vst    = kst + LW * 64;                                 // LW x 64 bf16

  const bf16_t* qrow = qb + ((size_t)bb * Sc + qbase) * Ec + h * Dc;
  const bf16_t* kmat = kb + (size_t)bb * Sc * Ec + h * Dc;
  const bf16_t* vmat = vb + (size_t)bb * Sc * Ec + h * Dc;

  int j0c = jbase < 0 ? 0 : jbase;       // clamped strip start
  int fskip = j0c - jbase;               // front rows not covered (masked later)
  int ravail = Sc - j0c;                 // rows before OOB (TDM zero-fills past)

  // zero-fill the uncovered front rows (keeps P*V exact: p=0 there)
  for (int i = tid; i < fskip * Dc; i += 128) { kst[i] = (bf16_t)0.f; vst[i] = (bf16_t)0.f; }

  // issue both strip DMAs from wave 0 (EXEC-independent, TENSORcnt-tracked)
  if (wave == 0) {
    tdm_load_2d_bf16((unsigned)(uintptr_t)(void*)(kst + fskip * Dc),
                     kmat + (size_t)j0c * Ec, Dc, (unsigned)ravail,
                     Dc, (unsigned)(LW - fskip), Ec);
    tdm_load_2d_bf16((unsigned)(uintptr_t)(void*)(vst + fskip * Dc),
                     vmat + (size_t)j0c * Ec, Dc, (unsigned)ravail,
                     Dc, (unsigned)(LW - fskip), Ec);
  }

  // ---- Phase 1a: global key tiles (overlaps with TDM) ----
  {
    int gt = wave;
    v8f c = {};
    for (int ks = 0; ks < Dc; ks += 32) {
      v16bf a = load_a_frag(qrow + ks, Ec, lane);
      v16bf b = load_bt_frag(kmat + (size_t)(gt * 16) * Ec + ks, Ec, lane);
      c = wmma_bf16(a, b, c);
    }
    int n = gt * 16 + (lane & 15);
    int mo = (lane >> 4) * 8;
#pragma unroll
    for (int r = 0; r < 8; ++r) s_glob[(mo + r) * 64 + n] = c[r];
  }
  if (wave == 0) __builtin_amdgcn_s_wait_tensorcnt(0);
  __syncthreads();

  // ---- Phase 1b: local strip from LDS ----
  for (int lt = wave; lt < LW / 16; lt += 4) {
    v8f c = {};
    int nn = lane & 15, kk = (lane >> 4) * 16;
    for (int ks = 0; ks < Dc; ks += 32) {
      v16bf a = load_a_frag(qrow + ks, Ec, lane);
      const bf16_t* r = kst + (size_t)(lt * 16 + nn) * Dc + ks + kk;
      v16bf b;
#pragma unroll
      for (int t = 0; t < 16; ++t) b[t] = r[t];
      c = wmma_bf16(a, b, c);
    }
    int mo = (lane >> 4) * 8;
    int j = jbase + lt * 16 + nn;
#pragma unroll
    for (int r = 0; r < 8; ++r) {
      int m = mo + r;
      int qa = qbase + m;
      int cdx = j - qa + Wc;
      bool valid = (j >= 0) && (j < Sc) && (cdx >= 0) && (cdx <= 2 * Wc);
      s_loc[m * LW + lt * 16 + nn] = valid ? c[r] : NEGINF;
    }
  }
  __syncthreads();

  // ---- Phase 2: softmax over 64 + 544 cols, 8 threads per row ----
  {
    int row = tid >> 3, t8 = tid & 7;
    float mx = NEGINF;
    for (int i = t8; i < 64; i += 8) mx = fmaxf(mx, s_glob[row * 64 + i]);
    for (int i = t8; i < LW; i += 8) mx = fmaxf(mx, s_loc[row * LW + i]);
    mx = fmaxf(mx, __shfl_xor(mx, 1));
    mx = fmaxf(mx, __shfl_xor(mx, 2));
    mx = fmaxf(mx, __shfl_xor(mx, 4));
    float sum = 0.f;
    for (int i = t8; i < 64; i += 8) sum += __expf(s_glob[row * 64 + i] - mx);
    for (int i = t8; i < LW; i += 8) sum += __expf(s_loc[row * LW + i] - mx);
    sum += __shfl_xor(sum, 1);
    sum += __shfl_xor(sum, 2);
    sum += __shfl_xor(sum, 4);
    float inv = 1.f / sum;
    int   qa  = qbase + row;
    float om  = (qa < Gc) ? 0.f : 1.f;   // is_index_global_attn output zeroing
    float* op = out_probs + (((size_t)bb * Sc + qa) * Hc + h) * PCOLS;
    for (int i = t8; i < 64; i += 8) {
      float p = __expf(s_glob[row * 64 + i] - mx) * inv;
      p_glob[row * 64 + i] = (bf16_t)p;
      op[i] = p * om;
    }
    for (int i = t8; i < LW; i += 8) {
      float p = __expf(s_loc[row * LW + i] - mx) * inv;
      p_loc[row * LW + i] = (bf16_t)p;
      int cdx = i - row;
      if (cdx >= 0 && cdx <= 2 * Wc) op[Gc + cdx] = p * om;
    }
  }
  __syncthreads();

  // ---- Phase 3: P x V, wave w owns d-tile w; V strip from LDS ----
  {
    int dbase = wave * 16;
    int nn = lane & 15, kk = (lane >> 4) * 16, mo = (lane >> 4) * 8;
    const bf16_t* vcol = vmat + dbase;
    v8f c = {};
    for (int kc = 0; kc < 2; ++kc) {     // global keys 0..63 (direct)
      v16bf a = load_a_frag(p_glob + kc * 32, 64, lane);
      v16bf b;
#pragma unroll
      for (int t = 0; t < 16; ++t)
        b[t] = vcol[(size_t)(kc * 32 + kk + t) * Ec + nn];
      c = wmma_bf16(a, b, c);
    }
    for (int kc = 0; kc < LW / 32; ++kc) {   // local strip (LDS)
      v16bf a = load_a_frag(p_loc + kc * 32, LW, lane);
      v16bf b;
#pragma unroll
      for (int t = 0; t < 16; ++t)
        b[t] = vst[(size_t)(kc * 32 + kk + t) * Dc + dbase + nn];
      c = wmma_bf16(a, b, c);
    }
#pragma unroll
    for (int r = 0; r < 8; ++r) {
      int qa = qbase + mo + r;
      out_att[((size_t)bb * Sc + qa) * Ec + h * Dc + dbase + nn] = c[r];
    }
  }
}

// ---------------- global attention: raw scores ------
__global__ void k_glob_scores(const bf16_t* __restrict__ qgb,
                              const bf16_t* __restrict__ kgb,
                              float* __restrict__ gp) {
  int bh = blockIdx.x, bb = bh >> 4, h = bh & 15;
  int mt = blockIdx.y, cg = blockIdx.z;
  int wave = threadIdx.x >> 5, lane = threadIdx.x & 31;

  const bf16_t* qrow = qgb + ((size_t)bb * Sc + mt * 16) * Ec + h * Dc;
  const bf16_t* kmat = kgb + (size_t)bb * Sc * Ec + h * Dc;
  float* out = gp + ((size_t)bh * Gc + mt * 16) * Sc;

  for (int nt = wave; nt < 16; nt += 4) {
    int n0 = cg * 256 + nt * 16;
    v8f c = {};
    for (int ks = 0; ks < Dc; ks += 32) {
      v16bf a = load_a_frag(qrow + ks, Ec, lane);
      v16bf b = load_bt_frag(kmat + (size_t)n0 * Ec + ks, Ec, lane);
      c = wmma_bf16(a, b, c);
    }
    int nn = n0 + (lane & 15), mo = (lane >> 4) * 8;
#pragma unroll
    for (int r = 0; r < 8; ++r) out[(size_t)(mo + r) * Sc + nn] = c[r];
  }
}

// ---------------- global attention: softmax + PV ----
__global__ void k_glob_soft_pv(float* __restrict__ gp,
                               const bf16_t* __restrict__ vgb,
                               float* __restrict__ out_att) {
  int bh = blockIdx.x, bb = bh >> 4, h = bh & 15, mt = blockIdx.y;
  int tid = threadIdx.x, wave = tid >> 5, lane = tid & 31;

  extern __shared__ char smem[];
  bf16_t* p = (bf16_t*)smem;            // 16 x 4096 bf16
  float* grow = gp + ((size_t)bh * Gc + mt * 16) * Sc;

  {
    int row = tid >> 4, t16 = tid & 15;
    float mx = NEGINF;
    for (int i = t16; i < Sc; i += 16) mx = fmaxf(mx, grow[(size_t)row * Sc + i]);
    mx = fmaxf(mx, __shfl_xor(mx, 1));
    mx = fmaxf(mx, __shfl_xor(mx, 2));
    mx = fmaxf(mx, __shfl_xor(mx, 4));
    mx = fmaxf(mx, __shfl_xor(mx, 8));
    float sum = 0.f;
    for (int i = t16; i < Sc; i += 16) sum += __expf(grow[(size_t)row * Sc + i] - mx);
    sum += __shfl_xor(sum, 1);
    sum += __shfl_xor(sum, 2);
    sum += __shfl_xor(sum, 4);
    sum += __shfl_xor(sum, 8);
    float inv = 1.f / sum;
    for (int i = t16; i < Sc; i += 16) {
      float pv = __expf(grow[(size_t)row * Sc + i] - mx) * inv;
      grow[(size_t)row * Sc + i] = pv;
      p[(size_t)row * Sc + i] = (bf16_t)pv;
    }
  }
  __syncthreads();

  if (wave < 4) {
    int dbase = wave * 16;
    const bf16_t* vcol = vgb + (size_t)bb * Sc * Ec + h * Dc + dbase;
    int nn = lane & 15, kk = (lane >> 4) * 16, mo = (lane >> 4) * 8;
    v8f c = {};
    for (int kc = 0; kc < Sc / 32; ++kc) {
      v16bf a = load_a_frag(p + kc * 32, Sc, lane);
      v16bf b;
#pragma unroll
      for (int t = 0; t < 16; ++t)
        b[t] = vcol[(size_t)(kc * 32 + kk + t) * Ec + nn];
      c = wmma_bf16(a, b, c);
    }
#pragma unroll
    for (int r = 0; r < 8; ++r) {
      int g = mt * 16 + mo + r;
      out_att[((size_t)bb * Sc + g) * Ec + h * Dc + dbase + nn] = c[r];
    }
  }
}

// ---------------- launcher --------------------------

extern "C" void kernel_launch(void* const* d_in, const int* in_sizes, int n_in,
                              void* d_out, int out_size, void* d_ws, size_t ws_size,
                              hipStream_t stream) {
  (void)in_sizes; (void)n_in; (void)out_size; (void)ws_size;

  const float* hs  = (const float*)d_in[0];
  const float* Wq  = (const float*)d_in[2];
  const float* bq  = (const float*)d_in[3];
  const float* Wk  = (const float*)d_in[4];
  const float* bk  = (const float*)d_in[5];
  const float* Wv  = (const float*)d_in[6];
  const float* bv  = (const float*)d_in[7];
  const float* Wqg = (const float*)d_in[8];
  const float* bqg = (const float*)d_in[9];
  const float* Wkg = (const float*)d_in[10];
  const float* bkg = (const float*)d_in[11];
  const float* Wvg = (const float*)d_in[12];
  const float* bvg = (const float*)d_in[13];

  char* ws = (char*)d_ws;
  const size_t HS_N = (size_t)Bc * Sc * Ec;
  const size_t W_N  = (size_t)Ec * Ec;
  bf16_t* hsb  = (bf16_t*)ws;
  bf16_t* wtb  = (bf16_t*)(ws + HS_N * 2);
  bf16_t* proj = (bf16_t*)(ws + HS_N * 2 + 6 * W_N * 2);
  const size_t PSTR = HS_N;

  float* out       = (float*)d_out;
  float* out_att   = out;
  float* out_probs = out + HS_N;
  float* out_gp    = out_probs + (size_t)Bc * Sc * Hc * PCOLS;

  k_cvt_bf16<<<(int)((HS_N + 255) / 256), 256, 0, stream>>>(hs, hsb, (int)HS_N);
  const float* Ws6[6] = {Wq, Wk, Wv, Wqg, Wkg, Wvg};
  for (int i = 0; i < 6; ++i)
    k_cvt_transpose<<<(int)((W_N + 255) / 256), 256, 0, stream>>>(Ws6[i], wtb + (size_t)i * W_N);

  GemmArgs ga;
  ga.bias[0] = bq;  ga.bias[1] = bk;  ga.bias[2] = bv;
  ga.bias[3] = bqg; ga.bias[4] = bkg; ga.bias[5] = bvg;
  ga.scale[0] = 0.125f; ga.scale[1] = 1.f; ga.scale[2] = 1.f;
  ga.scale[3] = 0.125f; ga.scale[4] = 1.f; ga.scale[5] = 1.f;
  k_proj_gemm<<<dim3(512, 4, 6), 128, 0, stream>>>(hsb, wtb, proj, ga);

  const int smem2 = 16 * LW * 4 + 16 * 64 * 4 + 16 * LW * 2 + 16 * 64 * 2
                  + 2 * LW * Dc * 2;   // scores/probs + K/V strips = 197632
  hipFuncSetAttribute(reinterpret_cast<const void*>(k_sliding_attn),
                      hipFuncAttributeMaxDynamicSharedMemorySize, smem2);
  k_sliding_attn<<<dim3(Sc / 16, Hc, Bc), 128, smem2, stream>>>(
      proj + 0 * PSTR, proj + 1 * PSTR, proj + 2 * PSTR, out_att, out_probs);

  k_glob_scores<<<dim3(Bc * Hc, 4, 16), 128, 0, stream>>>(
      proj + 3 * PSTR, proj + 4 * PSTR, out_gp);
  const int smem3 = 16 * Sc * 2;
  hipFuncSetAttribute(reinterpret_cast<const void*>(k_glob_soft_pv),
                      hipFuncAttributeMaxDynamicSharedMemorySize, smem3);
  k_glob_soft_pv<<<dim3(Bc * Hc, 4), 256, smem3, stream>>>(
      out_gp, proj + 5 * PSTR, out_att);
}
